// BeMultiHeadAttention_68899865362563
// MI455X (gfx1250) — matrix operand order repaired and checked by reading, not compiled, not run on hardware
//
#include <hip/hip_runtime.h>

#define B_ 2
#define S_ 2048
#define D_ 1024
#define H_ 16
#define HD_ 64

typedef __bf16 v16bf __attribute__((ext_vector_type(16)));
typedef float  v8f   __attribute__((ext_vector_type(8)));

__device__ inline v8f wmma_bf16(v16bf a, v16bf b, v8f c) {
    return __builtin_amdgcn_wmma_f32_16x16x32_bf16(false, a, false, b, (short)0, c, false, false);
}

// A fragment 16x32 (MxK), source row-major [m][k], row stride ld (elements).
// lane<16: m=lane, k in {0..7, 16..23}; lane>=16: m=lane-16, k in {8..15, 24..31}
__device__ inline v16bf load_a_frag(const __bf16* base, int ld, int lane) {
    int m  = lane & 15;
    int kb = (lane < 16) ? 0 : 8;
    const __bf16* r = base + (size_t)m * ld + kb;
    v16bf a;
#pragma unroll
    for (int j = 0; j < 8; ++j) { a[j] = r[j]; a[8 + j] = r[16 + j]; }
    return a;
}

// B fragment 32x16 (KxN), source N-major [n][k], row stride ld (elements).
// lane<16: n=lane, k in {0..7, 16..23}; lane>=16: n=lane-16, k in {8..15, 24..31}
__device__ inline v16bf load_b_frag(const __bf16* baseT, int ld, int lane) {
    int n  = lane & 15;
    int kb = (lane < 16) ? 0 : 8;
    const __bf16* r = baseT + (size_t)n * ld + kb;
    v16bf b;
#pragma unroll
    for (int j = 0; j < 8; ++j) { b[j] = r[j]; b[8 + j] = r[16 + j]; }
    return b;
}

// LDS byte offset of a __shared__ object: generic (flat) LDS addresses carry the
// aperture in the high 32 bits and the LDS offset in addr[31:0] (ISA 10.2).
__device__ inline unsigned lds_off(const void* p) {
    return (unsigned)(unsigned long long)(uintptr_t)p;
}

// Async global->LDS copy of 16 bytes per lane (tracked by ASYNCcnt).
__device__ inline void async_copy_b128(unsigned lds_byte, unsigned long long gaddr) {
    asm volatile("global_load_async_to_lds_b128 %0, %1, off"
                 :: "v"(lds_byte), "v"(gaddr) : "memory");
}

// ---------------- Kernel 1: per-head QKV projection -> bf16 staging ----------------
// Q stored pre-scaled by log2(e)/sqrt(HD); V stored transposed [d][t].
__global__ __launch_bounds__(128) void qkv_kernel(
    const float* __restrict__ X,
    const float* __restrict__ Wq, const float* __restrict__ bq,
    const float* __restrict__ Wk, const float* __restrict__ bk,
    const float* __restrict__ Wv, const float* __restrict__ bv,
    __bf16* __restrict__ Qs, __bf16* __restrict__ Ks, __bf16* __restrict__ Vt) {
    __shared__ __bf16 WT[3][64][64];   // [proj][e][d]  (transposed weights)
    __shared__ float  bias[3][64];

    const int tid = threadIdx.x;
    const int bh  = blockIdx.x;        // b*H + h
    const int b   = bh / H_, h = bh % H_;

    const float* Ws[3]  = {Wq, Wk, Wv};
    const float* bsp[3] = {bq, bk, bv};
#pragma unroll
    for (int p = 0; p < 3; ++p) {
        const float* W = Ws[p] + (size_t)h * 64 * 64;   // [d][e]
        for (int i = tid; i < 64 * 64; i += 128) {
            int d = i >> 6, e = i & 63;
            WT[p][e][d] = (__bf16)W[i];
        }
        if (tid < 64) bias[p][tid] = bsp[p][h * 64 + tid];
    }
    __syncthreads();

    const int w = tid >> 5, lane = tid & 31;
    const int s0 = blockIdx.y * 64 + w * 16;
    const int m = lane & 15, kb = (lane < 16) ? 0 : 8;
    const int n = lane & 15, g = lane >> 4;

    // A fragments from X (fp32 -> bf16), k = d in [0,64)
    const float* xr = X + ((size_t)b * S_ + (s0 + m)) * D_ + h * 64;
    v16bf a0, a1;
#pragma unroll
    for (int j = 0; j < 8; ++j) {
        a0[j]     = (__bf16)xr[kb + j];
        a0[8 + j] = (__bf16)xr[kb + 16 + j];
        a1[j]     = (__bf16)xr[32 + kb + j];
        a1[8 + j] = (__bf16)xr[32 + kb + 16 + j];
    }

    const float QSCALE = 0.125f * 1.44269504f;  // 1/sqrt(64) * log2(e)
#pragma unroll
    for (int p = 0; p < 3; ++p) {
#pragma unroll
        for (int nt = 0; nt < 4; ++nt) {
            v16bf b0 = load_b_frag(&WT[p][nt * 16][0],  64, lane);
            v16bf b1 = load_b_frag(&WT[p][nt * 16][32], 64, lane);
            v8f c = {};
            c = wmma_bf16(a0, b0, c);
            c = wmma_bf16(a1, b1, c);
            const int e  = nt * 16 + n;
            const float be = bias[p][e];
#pragma unroll
            for (int r = 0; r < 8; ++r) {
                float v = c[r] + be;
                int srow = s0 + r + 8 * g;
                if (p == 0)      Qs[((size_t)bh * S_ + srow) * 64 + e] = (__bf16)(v * QSCALE);
                else if (p == 1) Ks[((size_t)bh * S_ + srow) * 64 + e] = (__bf16)v;
                else             Vt[((size_t)bh * 64 + e) * S_ + srow] = (__bf16)v;
            }
        }
    }
}

// ---------------- Kernel 2: flash attention (S^T = K*Q^T, O^T = V^T*P^T) ----------------
// K/V blocks are streamed cooperatively into LDS with async copies, double-buffered.
__global__ __launch_bounds__(128) void attn_kernel(
    const __bf16* __restrict__ Qs, const __bf16* __restrict__ Ks,
    const __bf16* __restrict__ Vt, __bf16* __restrict__ Aout) {
    __shared__ __bf16 Kb[2][32][64];  // [buf][t][d]  4 KB each
    __shared__ __bf16 Vb[2][64][32];  // [buf][d][t]  4 KB each
    __shared__ __bf16 P[4][16][32];   // per-wave P^T bounce: [s][t]

    const int tid = threadIdx.x, w = tid >> 5, lane = tid & 31;
    const int bh = blockIdx.x;
    const int b  = bh / H_, h = bh % H_;
    const int s0 = blockIdx.y * 64 + w * 16;
    const int n  = lane & 15, g = lane >> 4;

    const __bf16* Qh = Qs + (size_t)bh * S_ * 64;
    const __bf16* Kh = Ks + (size_t)bh * S_ * 64;
    const __bf16* Vh = Vt + (size_t)bh * 64 * S_;

    const unsigned kl0 = lds_off(&Kb[0][0][0]);
    const unsigned vl0 = lds_off(&Vb[0][0][0]);
    const int vd = tid >> 2;                 // V row (d) handled by this thread
    const unsigned vtb = (tid & 3) * 16;     // byte offset within the 64B row chunk

    // Issue cooperative async fill of buffer `bufn` with the 32-key block at t0n.
    // Per thread: 2x16B of K (contiguous) + 2x16B of V (row-strided). 8 KB per WG.
    auto fill = [&](int bufn, int t0n) {
        unsigned long long kg = (unsigned long long)(uintptr_t)(Kh + (size_t)t0n * 64)
                                + (unsigned)tid * 16u;
        unsigned kl = kl0 + (unsigned)bufn * 4096u + (unsigned)tid * 16u;
        async_copy_b128(kl, kg);
        async_copy_b128(kl + 2048u, kg + 2048ull);
#pragma unroll
        for (int j = 0; j < 2; ++j) {
            int c = tid + j * 128;
            int d = c >> 2;
            unsigned long long vg = (unsigned long long)(uintptr_t)(Vh + (size_t)d * S_ + t0n)
                                    + vtb;
            unsigned vl = vl0 + (unsigned)bufn * 4096u + (unsigned)c * 16u;
            async_copy_b128(vl, vg);
        }
        (void)vd;
    };

    // Q as B fragments (N-major [s][d] == row-major Qs), k = d
    v16bf qb0 = load_b_frag(Qh + (size_t)s0 * 64,      64, lane);
    v16bf qb1 = load_b_frag(Qh + (size_t)s0 * 64 + 32, 64, lane);

    v8f o0 = {}, o1 = {}, o2 = {}, o3 = {};
    float mi = -3.0e38f, li = 0.0f;

    const int NIT = S_ / 32;
    fill(0, 0);
    for (int it = 0; it < NIT; ++it) {
        const int buf = it & 1;
        if (it + 1 < NIT) {
            fill(buf ^ 1, (it + 1) * 32);
            // async loads complete in order: <=4 outstanding => current buffer resident
            asm volatile("s_wait_asynccnt 0x4" ::: "memory");
        } else {
            asm volatile("s_wait_asynccnt 0x0" ::: "memory");
        }
        __syncthreads();

        // K as A fragments from LDS: tile tt covers keys tt*16..+15, k = d
        const __bf16* Kl = &Kb[buf][0][0];
        v16bf ka00 = load_a_frag(Kl,                64, lane);
        v16bf ka01 = load_a_frag(Kl + 32,           64, lane);
        v16bf ka10 = load_a_frag(Kl + 16 * 64,      64, lane);
        v16bf ka11 = load_a_frag(Kl + 16 * 64 + 32, 64, lane);

        v8f st0 = {}, st1 = {};
        st0 = wmma_bf16(ka00, qb0, st0);
        st0 = wmma_bf16(ka01, qb1, st0);
        st1 = wmma_bf16(ka10, qb0, st1);
        st1 = wmma_bf16(ka11, qb1, st1);
        // st* are S^T tiles (log2 domain): lane holds s=n, t = tt*16 + r + 8g

        float bm = -3.0e38f;
#pragma unroll
        for (int r = 0; r < 8; ++r) { bm = fmaxf(bm, st0[r]); bm = fmaxf(bm, st1[r]); }
        bm = fmaxf(bm, __shfl_xor(bm, 16, 32));
        float mnew = fmaxf(mi, bm);
        float corr = exp2f(mi - mnew);

        float p0[8], p1[8], rs = 0.0f;
#pragma unroll
        for (int r = 0; r < 8; ++r) {
            p0[r] = exp2f(st0[r] - mnew); rs += p0[r];
            p1[r] = exp2f(st1[r] - mnew); rs += p1[r];
        }
        rs += __shfl_xor(rs, 16, 32);
        li = li * corr + rs;
        mi = mnew;
#pragma unroll
        for (int r = 0; r < 8; ++r) { o0[r] *= corr; o1[r] *= corr; o2[r] *= corr; o3[r] *= corr; }

        // P^T -> LDS [s][t]
#pragma unroll
        for (int r = 0; r < 8; ++r) {
            P[w][n][r + 8 * g]      = (__bf16)p0[r];
            P[w][n][16 + r + 8 * g] = (__bf16)p1[r];
        }
        // P^T as B fragment (N-major [s][t]), k = t in [0,32)
        v16bf pb = load_b_frag(&P[w][0][0], 32, lane);

        // V^T as A fragments from LDS: tile dt covers d = dt*16.., k = t
        v16bf va0 = load_a_frag(&Vb[buf][ 0][0], 32, lane);
        v16bf va1 = load_a_frag(&Vb[buf][16][0], 32, lane);
        v16bf va2 = load_a_frag(&Vb[buf][32][0], 32, lane);
        v16bf va3 = load_a_frag(&Vb[buf][48][0], 32, lane);
        o0 = wmma_bf16(va0, pb, o0);
        o1 = wmma_bf16(va1, pb, o1);
        o2 = wmma_bf16(va2, pb, o2);
        o3 = wmma_bf16(va3, pb, o3);

        __syncthreads();   // protect buf from next iteration's prefetch
    }

    const float inv = 1.0f / li;
    const int srow = s0 + n;
    __bf16* outp = Aout + ((size_t)b * S_ + srow) * D_ + h * 64;
#pragma unroll
    for (int r = 0; r < 8; ++r) {
        outp[ 0 + 8 * g + r] = (__bf16)(o0[r] * inv);
        outp[16 + 8 * g + r] = (__bf16)(o1[r] * inv);
        outp[32 + 8 * g + r] = (__bf16)(o2[r] * inv);
        outp[48 + 8 * g + r] = (__bf16)(o3[r] * inv);
    }
}

// ---------------- Kernel 3: output projection Y = Aout @ Wo + bo ----------------
__global__ __launch_bounds__(128) void proj_kernel(
    const __bf16* __restrict__ Aout, const float* __restrict__ Wo,
    const float* __restrict__ bo, float* __restrict__ Y) {
    __shared__ __bf16 WT[64][32];   // Wo chunk transposed: [n][k]

    const int tid = threadIdx.x, w = tid >> 5, lane = tid & 31;
    const int m0 = blockIdx.x * 64 + w * 16;
    const int n0 = blockIdx.y * 64;
    const int n = lane & 15, g = lane >> 4;

    v8f acc[4] = {{}, {}, {}, {}};
    for (int k0 = 0; k0 < D_; k0 += 32) {
        __syncthreads();
        for (int i = tid; i < 64 * 32; i += 128) {
            int kk = i >> 6, nn = i & 63;
            WT[nn][kk] = (__bf16)Wo[(size_t)(k0 + kk) * D_ + n0 + nn];
        }
        __syncthreads();
        v16bf af = load_a_frag(Aout + (size_t)m0 * D_ + k0, D_, lane);
#pragma unroll
        for (int nt = 0; nt < 4; ++nt) {
            v16bf bf = load_b_frag(&WT[nt * 16][0], 32, lane);
            acc[nt] = wmma_bf16(af, bf, acc[nt]);
        }
    }
#pragma unroll
    for (int nt = 0; nt < 4; ++nt) {
        const int e = n0 + nt * 16 + n;
        const float be = bo[e];
#pragma unroll
        for (int r = 0; r < 8; ++r) {
            Y[((size_t)m0 + r + 8 * g) * D_ + e] = acc[nt][r] + be;
        }
    }
}

extern "C" void kernel_launch(void* const* d_in, const int* in_sizes, int n_in,
                              void* d_out, int out_size, void* d_ws, size_t ws_size,
                              hipStream_t stream) {
    const float* X  = (const float*)d_in[0];
    const float* Wq = (const float*)d_in[1];
    const float* bq = (const float*)d_in[2];
    const float* Wk = (const float*)d_in[3];
    const float* bk = (const float*)d_in[4];
    const float* Wv = (const float*)d_in[5];
    const float* bv = (const float*)d_in[6];
    const float* Wo = (const float*)d_in[7];
    const float* bo = (const float*)d_in[8];
    float* Y = (float*)d_out;

    const size_t perTensor = (size_t)B_ * H_ * S_ * HD_;   // 4M bf16 elements
    __bf16* Qs   = (__bf16*)d_ws;
    __bf16* Ks   = Qs + perTensor;
    __bf16* Vt   = Ks + perTensor;
    __bf16* Aout = Vt + perTensor;

    qkv_kernel<<<dim3(B_ * H_, S_ / 64), 128, 0, stream>>>(X, Wq, bq, Wk, bk, Wv, bv, Qs, Ks, Vt);
    attn_kernel<<<dim3(B_ * H_, S_ / 64), 128, 0, stream>>>(Qs, Ks, Vt, Aout);
    proj_kernel<<<dim3((B_ * S_) / 64, D_ / 64), 128, 0, stream>>>(Aout, Wo, bo, Y);
}
